// _DINOv3Attention_66151086293413
// MI455X (gfx1250) — matrix operand order repaired and checked by reading, not compiled, not run on hardware
//
#include <hip/hip_runtime.h>

typedef _Float16 half8  __attribute__((ext_vector_type(8)));
typedef _Float16 half16 __attribute__((ext_vector_type(16)));
typedef float    f32x8  __attribute__((ext_vector_type(8)));

#define DIM    1024
#define NHEADS 16
#define HDIM   64
#define BATCH  2
#define NTOK   2048
#define MROWS  (BATCH * NTOK)   // 4096
#define ATT_SCALE 0.125f        // 64^-0.5

// padded LDS row strides (in halves) to spread banks: gcd(stride/2,64) small
#define KV_STRIDE 72            // 32x64 tile rows padded 64 -> 72 (144B)
#define P_STRIDE  40            // 16x32 tile rows padded 32 -> 40 (80B)

static __device__ __forceinline__ f32x8 wmma16x16x32(half16 a, half16 b, f32x8 c) {
  // D = A(16x32) * B(32x16) + C, f32 accumulate
  return __builtin_amdgcn_wmma_f32_16x16x32_f16(false, a, false, b, (short)0, c,
                                                false, false);
}

// A-fragment loader: lane holds row (lane&15); half e at K=(lane>>4)*16+e.
// p points at the first chunk.
static __device__ __forceinline__ half16 load_a_frag(const _Float16* p) {
  half8 lo = *(const half8*)(p);
  half8 hi = *(const half8*)(p + 8);
  half16 r;
#pragma unroll
  for (int i = 0; i < 8; ++i) { r[i] = lo[i]; r[i + 8] = hi[i]; }
  return r;
}

// B-fragment loader from 16B-aligned memory (two half8 chunks, contiguous)
static __device__ __forceinline__ half16 load_b_frag16(const _Float16* p) {
  half8 lo = *(const half8*)(p);
  half8 hi = *(const half8*)(p + 8);
  half16 r;
#pragma unroll
  for (int i = 0; i < 8; ++i) { r[i] = lo[i]; r[i + 8] = hi[i]; }
  return r;
}

// ---- CDNA5 async global->LDS copy (ASYNCcnt-tracked, no VGPR round trip)
static __device__ __forceinline__ void async_load_b128(unsigned lds_off,
                                                       const void* gaddr) {
  asm volatile("global_load_async_to_lds_b128 %0, %1, off"
               ::"v"(lds_off), "v"(gaddr) : "memory");
}
static __device__ __forceinline__ void wait_async0() {
  asm volatile("s_wait_asynccnt 0x0" ::: "memory");
}

// ---------------------------------------------------------------- conversions
__global__ __launch_bounds__(256) void cvt_f32_to_f16_kernel(
    const float* __restrict__ src, _Float16* __restrict__ dst, int n) {
  int i = blockIdx.x * blockDim.x + threadIdx.x;
  if (i < n) dst[i] = (_Float16)src[i];
}

// W[k][n] (f32, 1024x1024) -> Wt[n][k] (f16), LDS-tiled 32x32 transpose
__global__ __launch_bounds__(256) void transpose_cvt_kernel(
    const float* __restrict__ src, _Float16* __restrict__ dst) {
  __shared__ float tile[32][33];
  const int k0 = blockIdx.x * 32;
  const int n0 = blockIdx.y * 32;
  const int tx = threadIdx.x;   // 0..31
  const int ty = threadIdx.y;   // 0..7
#pragma unroll
  for (int i = ty; i < 32; i += 8)
    tile[i][tx] = src[(size_t)(k0 + i) * DIM + n0 + tx];
  __syncthreads();
#pragma unroll
  for (int i = ty; i < 32; i += 8)
    dst[(size_t)(n0 + i) * DIM + k0 + tx] = (_Float16)tile[tx][i];
}

// ---------------------------------------------------------------- WMMA GEMM
// out[M,N] = A[M,K] @ Wt[N,K]^T + bias.  One wave per 32x64 output tile:
// 2 A-fragments reused across 4 B-fragments -> 8 WMMAs per 32-deep k-step.
template <typename OutT>
__global__ __launch_bounds__(128) void wmma_gemm_kernel(
    const _Float16* __restrict__ A, const _Float16* __restrict__ Wt,
    const float* __restrict__ bias, OutT* __restrict__ out,
    int M, int N, int K) {
  const int lane = threadIdx.x & 31;
  const int wave = threadIdx.x >> 5;
  const int llo = lane & 15, lhi = lane >> 4;
  const int waveId = blockIdx.x * (blockDim.x >> 5) + wave;
  const int nTiles = N >> 6;
  const int mBase = (waveId / nTiles) << 5;   // 32 rows per wave
  const int nBase = (waveId % nTiles) << 6;   // 64 cols per wave
  if (mBase >= M) return;  // wave-uniform

  f32x8 acc[2][4];
#pragma unroll
  for (int i = 0; i < 2; ++i)
#pragma unroll
    for (int j = 0; j < 4; ++j)
#pragma unroll
      for (int r = 0; r < 8; ++r) acc[i][j][r] = 0.0f;

  const _Float16* arow0 = A + (size_t)(mBase + llo) * K + lhi * 16;
  const _Float16* arow1 = arow0 + (size_t)16 * K;
#pragma unroll 2
  for (int kk = 0; kk < K; kk += 32) {
    half16 a0 = load_a_frag(arow0 + kk);
    half16 a1 = load_a_frag(arow1 + kk);
#pragma unroll
    for (int j = 0; j < 4; ++j) {
      const _Float16* bp =
          Wt + (size_t)(nBase + j * 16 + llo) * K + kk + lhi * 16;
      half16 b = *(const half16*)bp;  // 16 contiguous K halves (32B aligned)
      acc[0][j] = wmma16x16x32(a0, b, acc[0][j]);
      acc[1][j] = wmma16x16x32(a1, b, acc[1][j]);
    }
  }

#pragma unroll
  for (int j = 0; j < 4; ++j) {
    const float bv = bias ? bias[nBase + j * 16 + llo] : 0.0f;
#pragma unroll
    for (int i = 0; i < 2; ++i)
#pragma unroll
      for (int r = 0; r < 8; ++r) {
        float v = acc[i][j][r] + bv;
        out[(size_t)(mBase + i * 16 + r + 8 * lhi) * N + nBase + j * 16 + llo] =
            (OutT)v;
      }
  }
}

// ---------------------------------------------------------------- attention
// Flash-style: 4 waves/block; wave owns 16 query rows of one head; loop over
// key tiles of 32. q/k/v are f16 [MROWS][DIM] with column = h*64 + d.
__global__ __launch_bounds__(128) void attn_kernel(
    const _Float16* __restrict__ q16, const _Float16* __restrict__ k16,
    const _Float16* __restrict__ v16, _Float16* __restrict__ o16) {
  __shared__ __align__(32) _Float16 lds_k[32 * KV_STRIDE];
  __shared__ __align__(32) _Float16 lds_v[32 * KV_STRIDE];
  __shared__ __align__(32) _Float16 lds_p[4 * 16 * P_STRIDE];

  const int tid = threadIdx.x;
  const int lane = tid & 31;
  const int wave = tid >> 5;
  const int llo = lane & 15, lhi = lane >> 4;

  const int bh = blockIdx.x >> 5;    // b*16 + h
  const int qBlk = blockIdx.x & 31;  // 64-row query block
  const int b = bh >> 4;
  const int h = bh & 15;
  const int qBase = qBlk * 64 + wave * 16;
  const int rowQ0 = b * NTOK + qBase;
  const int colH = h * HDIM;

  // Q fragments (K-dim 64 -> 2 chunks of 32), loaded once
  half16 qf[2];
#pragma unroll
  for (int c = 0; c < 2; ++c)
    qf[c] = load_a_frag(q16 + (size_t)(rowQ0 + llo) * DIM + colH + c * 32 +
                        lhi * 16);

  f32x8 o[4];
  float mrow[8], lrow[8];
#pragma unroll
  for (int j = 0; j < 4; ++j)
#pragma unroll
    for (int r = 0; r < 8; ++r) o[j][r] = 0.0f;
#pragma unroll
  for (int r = 0; r < 8; ++r) { mrow[r] = -1e30f; lrow[r] = 0.0f; }

  _Float16* pt = lds_p + wave * (16 * P_STRIDE);

  // staging: 128 threads move a 32x64 tile (32B per thread per buffer)
  const int stageKey = tid >> 2;        // 0..31
  const int stageD = (tid & 3) * 16;    // 0,16,32,48
  const unsigned kOff =
      (unsigned)(uintptr_t)(lds_k + stageKey * KV_STRIDE + stageD);
  const unsigned vOff =
      (unsigned)(uintptr_t)(lds_v + stageKey * KV_STRIDE + stageD);
  const _Float16* gK =
      k16 + (size_t)(b * NTOK + stageKey) * DIM + colH + stageD;
  const _Float16* gV =
      v16 + (size_t)(b * NTOK + stageKey) * DIM + colH + stageD;

  for (int kb = 0; kb < NTOK; kb += 32) {
    __syncthreads();  // previous-iteration LDS reads done
    // CDNA5 async global->LDS staging (8 halves per async-b128)
    async_load_b128(kOff, gK + (size_t)kb * DIM);
    async_load_b128(kOff + 16, gK + (size_t)kb * DIM + 8);
    async_load_b128(vOff, gV + (size_t)kb * DIM);
    async_load_b128(vOff + 16, gV + (size_t)kb * DIM + 8);
    wait_async0();
    __syncthreads();

    // S = Q @ K^T  -> two 16x16 fragments (keys 0-15 and 16-31)
    f32x8 s[2];
#pragma unroll
    for (int t = 0; t < 2; ++t) {
#pragma unroll
      for (int r = 0; r < 8; ++r) s[t][r] = 0.0f;
#pragma unroll
      for (int c = 0; c < 2; ++c) {
        half16 bk = load_b_frag16(lds_k + (t * 16 + llo) * KV_STRIDE + c * 32 +
                                  lhi * 16);
        s[t] = wmma16x16x32(qf[c], bk, s[t]);
      }
    }

    // online softmax: rows live in 16-lane half-groups at element r
    float fac[8];
#pragma unroll
    for (int r = 0; r < 8; ++r) {
      float s0 = s[0][r] * ATT_SCALE, s1 = s[1][r] * ATT_SCALE;
      float rm = fmaxf(s0, s1);
      rm = fmaxf(rm, __shfl_xor(rm, 1, 32));
      rm = fmaxf(rm, __shfl_xor(rm, 2, 32));
      rm = fmaxf(rm, __shfl_xor(rm, 4, 32));
      rm = fmaxf(rm, __shfl_xor(rm, 8, 32));
      float mn = fmaxf(mrow[r], rm);
      float f = __expf(mrow[r] - mn);
      float p0 = __expf(s0 - mn);
      float p1 = __expf(s1 - mn);
      float rs = p0 + p1;
      rs += __shfl_xor(rs, 1, 32);
      rs += __shfl_xor(rs, 2, 32);
      rs += __shfl_xor(rs, 4, 32);
      rs += __shfl_xor(rs, 8, 32);
      lrow[r] = lrow[r] * f + rs;
      mrow[r] = mn;
      fac[r] = f;
      // C-layout -> LDS (row = r + 8*lhi, col = key)
      pt[(r + 8 * lhi) * P_STRIDE + llo] = (_Float16)p0;
      pt[(r + 8 * lhi) * P_STRIDE + 16 + llo] = (_Float16)p1;
    }
#pragma unroll
    for (int j = 0; j < 4; ++j)
#pragma unroll
      for (int r = 0; r < 8; ++r) o[j][r] *= fac[r];

    __syncthreads();  // P LDS writes visible / compiler ordering

    // P (16x32) as A-fragment; O += P @ V
    half16 pa = load_a_frag(pt + llo * P_STRIDE + lhi * 16);
#pragma unroll
    for (int j = 0; j < 4; ++j) {
      half16 bv;
#pragma unroll
      for (int e = 0; e < 16; ++e)
        bv[e] = lds_v[(lhi * 16 + e) * KV_STRIDE + j * 16 + llo];
      o[j] = wmma16x16x32(pa, bv, o[j]);
    }
  }

  // normalize rows and store f16 to [b, n, h*64 + d]
#pragma unroll
  for (int j = 0; j < 4; ++j) {
#pragma unroll
    for (int r = 0; r < 8; ++r) {
      float v = o[j][r] / lrow[r];
      o16[(size_t)(rowQ0 + r + 8 * lhi) * DIM + colH + j * 16 + llo] =
          (_Float16)v;
    }
  }
}

// ---------------------------------------------------------------- launch
extern "C" void kernel_launch(void* const* d_in, const int* in_sizes, int n_in,
                              void* d_out, int out_size, void* d_ws,
                              size_t ws_size, hipStream_t stream) {
  (void)in_sizes; (void)n_in; (void)out_size; (void)ws_size;
  const float* x  = (const float*)d_in[0];
  const float* Wq = (const float*)d_in[1];
  const float* bq = (const float*)d_in[2];
  const float* Wk = (const float*)d_in[3];
  const float* Wv = (const float*)d_in[4];
  const float* bv = (const float*)d_in[5];
  const float* Wo = (const float*)d_in[6];
  const float* bo = (const float*)d_in[7];
  float* out = (float*)d_out;

  char* ws = (char*)d_ws;
  const size_t XB = (size_t)MROWS * DIM * 2;  // 8 MB
  const size_t WB = (size_t)DIM * DIM * 2;    // 2 MB
  _Float16* x16 = (_Float16*)(ws);            // also reused as attn output
  _Float16* wqt = (_Float16*)(ws + XB);
  _Float16* wkt = (_Float16*)(ws + XB + WB);
  _Float16* wvt = (_Float16*)(ws + XB + 2 * WB);
  _Float16* wot = (_Float16*)(ws + XB + 3 * WB);
  _Float16* q16 = (_Float16*)(ws + XB + 4 * WB);
  _Float16* k16 = (_Float16*)(ws + 2 * XB + 4 * WB);
  _Float16* v16 = (_Float16*)(ws + 3 * XB + 4 * WB);
  _Float16* a16 = x16;  // attention output reuses x16 (x consumed by then)

  // 1) f32 -> f16 conversions
  {
    int n = MROWS * DIM;
    cvt_f32_to_f16_kernel<<<(n + 255) / 256, 256, 0, stream>>>(x, x16, n);
  }
  {
    dim3 g(DIM / 32, DIM / 32), blk(32, 8);
    transpose_cvt_kernel<<<g, blk, 0, stream>>>(Wq, wqt);
    transpose_cvt_kernel<<<g, blk, 0, stream>>>(Wk, wkt);
    transpose_cvt_kernel<<<g, blk, 0, stream>>>(Wv, wvt);
    transpose_cvt_kernel<<<g, blk, 0, stream>>>(Wo, wot);
  }

  // 2) QKV projections: (4096/32)*(1024/64) = 2048 waves -> 512 blocks x 128
  const int gemmBlocks = (MROWS / 32) * (DIM / 64) / 4;
  wmma_gemm_kernel<_Float16><<<gemmBlocks, 128, 0, stream>>>(
      x16, wqt, bq, q16, MROWS, DIM, DIM);
  wmma_gemm_kernel<_Float16><<<gemmBlocks, 128, 0, stream>>>(
      x16, wkt, (const float*)nullptr, k16, MROWS, DIM, DIM);
  wmma_gemm_kernel<_Float16><<<gemmBlocks, 128, 0, stream>>>(
      x16, wvt, bv, v16, MROWS, DIM, DIM);

  // 3) attention: B*H * (2048/64) = 1024 blocks x 128 threads
  attn_kernel<<<BATCH * NHEADS * (NTOK / 64), 128, 0, stream>>>(q16, k16, v16,
                                                                a16);

  // 4) output projection (f32 out + bias)
  wmma_gemm_kernel<float><<<gemmBlocks, 128, 0, stream>>>(a16, wot, bo, out,
                                                          MROWS, DIM, DIM);
}
